// InteractionHead_77309411760
// MI455X (gfx1250) — compile-verified, matrix operand order checked by reading, not run on hardware
//
#include <hip/hip_runtime.h>
#include <hip/hip_bf16.h>

// ---------------------------------------------------------------------------
// InteractionHead label assignment, MI455X (gfx1250, wave32).
//
// Shapes: B=8, N=128, M=32, C=600, P=N*N=16384.
// Output: [paired_idx (16384x2) as float | labels (8,16384,600) float] =
//         314.7 MB -> store-bandwidth bound (~13.5 us @ 23.3 TB/s).
// Strategy:
//   k1: precompute iouh[b,n,m]=IoU(boxes[b,n],boxes_h[b,m]) and
//       iouo[b,n,m]=IoU(boxes[b,n],boxes_o[b,m]) tables (2x128KB in d_ws),
//       plus the constant paired_idx output.
//   k2: one wave per (b, 16-pair tile). K=32 == M exactly, so the
//       match(16x32) @ onehot(32x16) einsum is ONE v_wmma_f32_16x16x32_f16
//       per 16x16 output tile (0/1 in f16 and sums<=32 in f32 are exact).
//       38 class tiles (600 padded to 608, tail masked). Non-temporal
//       stores keep the 315MB write-once stream out of L2.
// ---------------------------------------------------------------------------

typedef __attribute__((ext_vector_type(16))) _Float16 v16h;
typedef __attribute__((ext_vector_type(8)))  float    v8f;

namespace {
constexpr int Bn = 8, Nn = 128, Mn = 32, Cn = 600;
constexpr int Pn = Nn * Nn;              // 16384 pairs
constexpr int HUMAN = 49;
constexpr int NTILE = (Cn + 15) / 16;    // 38 class tiles (608 padded)
}

__device__ __forceinline__ float iou1(float4 a, float4 b) {
    float lx = fmaxf(a.x, b.x), ly = fmaxf(a.y, b.y);
    float rx = fminf(a.z, b.z), ry = fminf(a.w, b.w);
    float w = fmaxf(rx - lx, 0.0f), h = fmaxf(ry - ly, 0.0f);
    float inter = w * h;
    float aa = (a.z - a.x) * (a.w - a.y);
    float ab = (b.z - b.x) * (b.w - b.y);
    return inter / (aa + ab - inter);
}

// One thread per (b, n, m): fill both IoU tables; first 16384 threads also
// emit the constant paired_idx block of the output.
__global__ void iou_tables_kernel(const float* __restrict__ boxes,
                                  const float* __restrict__ boxes_h,
                                  const float* __restrict__ boxes_o,
                                  float* __restrict__ iouh,
                                  float* __restrict__ iouo,
                                  float* __restrict__ paired_out) {
    int tid = blockIdx.x * blockDim.x + threadIdx.x;   // 32768 threads
    int b = tid >> 12;            // /(N*M)
    int n = (tid >> 5) & (Nn - 1);
    int m = tid & (Mn - 1);

    float4 bn = *(const float4*)(boxes   + ((size_t)(b * Nn + n)) * 4);
    float4 bh = *(const float4*)(boxes_h + ((size_t)(b * Mn + m)) * 4);
    float4 bo = *(const float4*)(boxes_o + ((size_t)(b * Mn + m)) * 4);

    iouh[tid] = iou1(bn, bh);     // layout [b][n][m], stride matches tid
    iouo[tid] = iou1(bn, bo);

    if (tid < Pn) {               // paired_idx = (p/128, p%128) as floats
        paired_out[2 * tid + 0] = (float)(tid >> 7);
        paired_out[2 * tid + 1] = (float)(tid & (Nn - 1));
    }
}

// One wave per (b, 16-pair tile): 38 WMMAs, 304 NT stores (~38KB) each.
__global__ void labels_wmma_kernel(const float* __restrict__ iouh,
                                   const float* __restrict__ iouo,
                                   const int*   __restrict__ det_labels,
                                   const int*   __restrict__ hoi,
                                   float* __restrict__ out_labels) {
    const int lane  = threadIdx.x & 31;
    const int wave  = (blockIdx.x * blockDim.x + threadIdx.x) >> 5; // 0..8191
    const int b     = wave >> 10;          // 1024 pair-tiles per batch
    const int ptile = wave & 1023;
    const int h     = ptile >> 3;          // 8 tiles per h row: tiles never
    const int o0    = (ptile & 7) << 4;    //   cross an h boundary
    const int half  = lane >> 4;           // lane group (ISA K split)
    const int nrow  = lane & 15;
    const int kbase = half << 3;

    // --- Build A: 16x32 match bits in the ISA 16-bit A layout -------------
    // lane row M = nrow; vector elem j holds K = kbase + (j&7) + 16*(j>>3).
    const float* ihp  = iouh + (size_t)(b * Nn + h) * Mn;          // uniform
    const float* iop  = iouo + (size_t)(b * Nn + (o0 + nrow)) * Mn; // per row
    const int*   hoip = hoi + b * Mn;

    v16h amat;
    int  hm[16];                            // hoi cached for all 38 B builds
#pragma unroll
    for (int j = 0; j < 16; ++j) {
        int m = kbase + (j & 7) + ((j >> 3) << 4);
        float v = fminf(ihp[m], iop[m]);
        amat[j] = (v >= 0.5f) ? (_Float16)1.0f : (_Float16)0.0f;
        hm[j] = hoip[m];
    }

    // --- Per-row validity: (det_labels[h]==49) && (h != o) ----------------
    const bool hval = (det_labels[b * Nn + h] == HUMAN);
    float rowscale[8];
#pragma unroll
    for (int r = 0; r < 8; ++r) {
        int rowM = r + (half << 3);         // D row for VGPR r in this lane
        rowscale[r] = (hval && (h != (o0 + rowM))) ? 1.0f : 0.0f;
    }

    float* base = out_labels + ((size_t)b * Pn + (size_t)ptile * 16) * Cn;

    // --- 38 class tiles of 16: B = onehot(hoi) built in registers ---------
    for (int ct = 0; ct < NTILE; ++ct) {
        const int c = ct * 16 + nrow;       // B column N = lane&15 = class
        v16h bmat;
#pragma unroll
        for (int j = 0; j < 16; ++j)
            bmat[j] = (hm[j] == c) ? (_Float16)1.0f : (_Float16)0.0f;

        v8f acc = {};
        acc = __builtin_amdgcn_wmma_f32_16x16x32_f16(
            /*neg_a=*/false, amat, /*neg_b=*/false, bmat,
            /*c_mod=*/(short)0, acc, /*reuse_a=*/false, /*reuse_b=*/false);

        if (c < Cn) {                        // mask padded tail (600..607)
#pragma unroll
            for (int r = 0; r < 8; ++r) {
                int rowM = r + (half << 3);
                float v = fminf(acc[r], 1.0f) * rowscale[r];
                __builtin_nontemporal_store(v, base + (size_t)rowM * Cn + c);
            }
        }
    }
}

extern "C" void kernel_launch(void* const* d_in, const int* in_sizes, int n_in,
                              void* d_out, int out_size, void* d_ws, size_t ws_size,
                              hipStream_t stream) {
    const float* boxes      = (const float*)d_in[0];   // (B,N,4)
    const float* boxes_h    = (const float*)d_in[1];   // (B,M,4)
    const float* boxes_o    = (const float*)d_in[2];   // (B,M,4)
    const int*   det_labels = (const int*)d_in[3];     // (B,N)
    const int*   hoi        = (const int*)d_in[4];     // (B,M)

    float* out        = (float*)d_out;
    float* paired_out = out;                 // 16384*2 floats
    float* labels_out = out + 2 * Pn;        // (B,P,C) floats

    float* iouh = (float*)d_ws;              // B*N*M floats (128 KB)
    float* iouo = iouh + Bn * Nn * Mn;       // B*N*M floats (128 KB)

    // k1: 8*128*32 = 32768 threads
    iou_tables_kernel<<<(Bn * Nn * Mn) / 256, 256, 0, stream>>>(
        boxes, boxes_h, boxes_o, iouh, iouo, paired_out);

    // k2: 8192 waves = 8 waves/block * 1024 blocks
    labels_wmma_kernel<<<(Bn * (Pn / 16) * 32) / 256, 256, 0, stream>>>(
        iouh, iouo, det_labels, hoi, labels_out);
}
